// RGCNBasisLayer_18657337933972
// MI455X (gfx1250) — matrix operand order repaired and checked by reading, not compiled, not run on hardware
//
#include <hip/hip_runtime.h>
#include <hip/hip_bf16.h>

// ---------------------------------------------------------------------------
// RGCN basis layer for MI455X (gfx1250, wave32, WMMA, LDS-staged B).
//
//  Phase 1: Hb[n, o*8+b] = sum_i h[n,i] * weight[b,i,o]   (bf16 WMMA GEMM,
//           B strip staged in LDS and shared by all 8 waves of a block;
//           output stored feature-major so the edge gather is one b128/thread)
//  Phase 2: acc[dst] += norm_e * sum_b w_comp[etype_e,b] * Hb[src_e, :, b]
//  Phase 3: out = relu(acc + bias)
// ---------------------------------------------------------------------------

typedef __attribute__((ext_vector_type(16))) __bf16 v16bf;
typedef __attribute__((ext_vector_type(8)))  __bf16 v8bf;
typedef __attribute__((ext_vector_type(8)))  float  v8f;

#define N_NODES  50000
#define N_EDGES  500000
#define IN_FEAT  128
#define OUT_FEAT 128
#define NUM_RELS 32
#define NUM_BASES 8
#define CAT_COLS (NUM_BASES * OUT_FEAT)   // 1024
#define M_TILES  (N_NODES / 16)           // 3125
#define M_GROUPS ((M_TILES + 7) / 8)      // 391 (tail guarded)
#define SB_STRIDE 136                     // 128 + 8 pad: conflict-free ds_b128

// ---------------------------------------------------------------------------
// Kernel 1a: convert node features f32 -> bf16 (row-major [N, 128])
// ---------------------------------------------------------------------------
__global__ void rgcn_convert_h(const float* __restrict__ h,
                               __bf16* __restrict__ h_bf) {
    int i = blockIdx.x * blockDim.x + threadIdx.x;
    if (i < N_NODES * IN_FEAT) h_bf[i] = (__bf16)h[i];
}

// ---------------------------------------------------------------------------
// Kernel 1b: build transposed concatenated weight Wt[c][i], c = b*128 + o.
// Wt[c*128 + i] = weight[b, i, o].  Column-major-for-B layout: each WMMA
// B-fragment (two 8-value K runs of one column, 16 apart) is two 16B chunks.
// ---------------------------------------------------------------------------
__global__ void rgcn_pack_w(const float* __restrict__ weight,
                            __bf16* __restrict__ wt) {
    int idx = blockIdx.x * blockDim.x + threadIdx.x;   // over 1024*128
    if (idx >= CAT_COLS * IN_FEAT) return;
    int c = idx >> 7;          // 0..1023
    int i = idx & 127;         // 0..127
    int b = c >> 7;            // basis
    int o = c & 127;           // out feature
    wt[idx] = (__bf16)weight[b * IN_FEAT * OUT_FEAT + i * OUT_FEAT + o];
}

// ---------------------------------------------------------------------------
// Kernel 1c: zero the scatter accumulator (must happen every call: graph
// replay re-runs the whole launch and the harness does not re-poison).
// ---------------------------------------------------------------------------
__global__ void rgcn_zero_acc(float* __restrict__ acc) {
    int i = blockIdx.x * blockDim.x + threadIdx.x;
    if (i < N_NODES * OUT_FEAT) acc[i] = 0.0f;
}

// ---------------------------------------------------------------------------
// Kernel 2: WMMA GEMM  Hb = h_bf @ Wt^T   ([50000 x 128] @ [128 x 1024])
//
// Block = 8 waves sharing ONE 64-column strip of Wt (16 KB -> LDS once,
// 8x less global B traffic); each wave owns a 16-row m-tile of that strip.
// Per wave: 4 f32 accumulators, K=128 in 4 steps of 32, A preloaded for all
// of K from global, B fragments fed from LDS -> 16 v_wmma per wave with only
// short ds-latency dependencies.
// LDS columns padded to 136 bf16 (272B stride): per-lane bank advance of 4
// banks makes the 16B ds_load_b128 fragment reads conflict-free.
// ---------------------------------------------------------------------------
__global__ void __launch_bounds__(256)
rgcn_wmma_gemm(const __bf16* __restrict__ h_bf,
               const __bf16* __restrict__ wt,
               __bf16* __restrict__ hb) {
    __shared__ __bf16 sB[64 * SB_STRIDE];   // 17408 B

    const int lane   = threadIdx.x & 31;
    const int wave   = threadIdx.x >> 5;
    const int mgroup = blockIdx.x >> 4;     // 0..390
    const int nstrip = blockIdx.x & 15;     // 0..15 (64 cols each)
    const int c0     = nstrip * 64;

    // --- Cooperative stage of the 64x128 B strip into LDS (16 KB) ----------
    // 1024 16-byte chunks, 256 threads, 4 chunks each.
    #pragma unroll
    for (int t = threadIdx.x; t < 64 * 16; t += 256) {
        const int c  = t >> 4;     // strip-local column 0..63
        const int kc = t & 15;     // 8-element K chunk
        *(v8bf*)&sB[c * SB_STRIDE + kc * 8] =
            *(const v8bf*)(wt + (size_t)(c0 + c) * IN_FEAT + kc * 8);
    }
    __syncthreads();

    // Tail guard (3125 m-tiles, groups of 8): wave-uniform, after the barrier.
    const int mtile = mgroup * 8 + wave;
    if (mtile >= M_TILES) return;
    const int m0 = mtile * 16;

    // A-fragment addressing (16-bit A 16x32 layout):
    //  lanes 0-15 : M=lane,    K in {k0+0..7,  k0+16..23}
    //  lanes 16-31: M=lane-16, K in {k0+8..15, k0+24..31}
    const int mrow = m0 + (lane & 15);
    const int koff = (lane < 16) ? 0 : 8;
    const __bf16* arow = h_bf + (size_t)mrow * IN_FEAT;

    // B-fragment addressing (32x16 bf16 B):
    //  lanes 0-15 : N=lane,    K in {k0+0..7,  k0+16..23}
    //  lanes 16-31: N=lane-16, K in {k0+8..15, k0+24..31}
    const int ncol  = lane & 15;
    const int kbase = (lane < 16) ? 0 : 8;
    const __bf16* sbcol = &sB[ncol * SB_STRIDE + kbase];

    // Preload all A fragments for K = 0..127 (8 global b128 loads).
    v16bf A[4];
    #pragma unroll
    for (int k = 0; k < 4; ++k) {
        v8bf lo = *(const v8bf*)(arow + k * 32 + koff);
        v8bf hi = *(const v8bf*)(arow + k * 32 + koff + 16);
        #pragma unroll
        for (int i = 0; i < 8; ++i) { A[k][i] = lo[i]; A[k][8 + i] = hi[i]; }
    }

    v8f acc[4] = {};

    #pragma unroll
    for (int k = 0; k < 4; ++k) {
        #pragma unroll
        for (int n = 0; n < 4; ++n) {
            // Fragment from LDS as two aligned 16B ds loads (K chunks +0 and +16).
            const __bf16* p = sbcol + n * 16 * SB_STRIDE + k * 32;
            v8bf blo = *(const v8bf*)(p);
            v8bf bhi = *(const v8bf*)(p + 16);
            v16bf B;
            #pragma unroll
            for (int i = 0; i < 8; ++i) { B[i] = blo[i]; B[8 + i] = bhi[i]; }
            acc[n] = __builtin_amdgcn_wmma_f32_16x16x32_bf16(
                false, A[k], false, B, (short)0, acc[n], false, false);
        }
    }

    // D layout (32-bit C/D 16x16): VGPR r -> M = r + (lane<16 ? 0 : 8), N = lane&15.
    // Store feature-major: index = row*1024 + o*8 + b, with col = b*128 + o.
    const int drow = m0 + ((lane >> 4) << 3);
    #pragma unroll
    for (int n = 0; n < 4; ++n) {
        const int col = c0 + n * 16 + ncol;
        const int b   = col >> 7;
        const int o   = col & 127;
        #pragma unroll
        for (int r = 0; r < 8; ++r) {
            hb[(size_t)(drow + r) * CAT_COLS + o * NUM_BASES + b] = (__bf16)acc[n][r];
        }
    }
}

// ---------------------------------------------------------------------------
// Kernel 3: edge scatter. 128 threads per edge (thread = out feature o),
// 2 edges per 256-thread block. Feature-major Hb layout: each thread's 8
// basis values are one contiguous 16B load (global_load_b128); each edge's
// gather is a single contiguous 2KB chunk. Combine with w_comp[etype] in
// f32, scale by norm, atomic-add to acc[dst] (coalesced f32 atomics).
// ---------------------------------------------------------------------------
__global__ void __launch_bounds__(256)
rgcn_edge_scatter(const __bf16* __restrict__ hb,
                  const int*   __restrict__ src,
                  const int*   __restrict__ dst,
                  const int*   __restrict__ etype,
                  const float* __restrict__ norm,
                  const float* __restrict__ w_comp,
                  float*       __restrict__ acc) {
    const int e = blockIdx.x * 2 + (threadIdx.x >> 7);
    const int o = threadIdx.x & 127;
    if (e >= N_EDGES) return;

    const int   s  = src[e];
    const int   d  = dst[e];
    const int   r  = etype[e];
    const float nm = norm[e];
    const float* wc = w_comp + r * NUM_BASES;

    // One 16-byte load: 8 basis projections for this (src node, feature).
    v8bf hv = *(const v8bf*)(hb + (size_t)s * CAT_COLS + o * NUM_BASES);

    float m = 0.0f;
    #pragma unroll
    for (int b = 0; b < NUM_BASES; ++b) {
        m += wc[b] * (float)hv[b];
    }
    atomicAdd(&acc[d * OUT_FEAT + o], m * nm);
}

// ---------------------------------------------------------------------------
// Kernel 4: out = relu(acc + bias)
// ---------------------------------------------------------------------------
__global__ void rgcn_finalize(const float* __restrict__ acc,
                              const float* __restrict__ bias,
                              float* __restrict__ out) {
    int i = blockIdx.x * blockDim.x + threadIdx.x;
    if (i >= N_NODES * OUT_FEAT) return;
    float v = acc[i] + bias[i & 127];
    out[i] = v > 0.0f ? v : 0.0f;
}

// ---------------------------------------------------------------------------
// Host launcher
// ---------------------------------------------------------------------------
extern "C" void kernel_launch(void* const* d_in, const int* in_sizes, int n_in,
                              void* d_out, int out_size, void* d_ws, size_t ws_size,
                              hipStream_t stream) {
    const float* h       = (const float*)d_in[0];
    const int*   src     = (const int*)  d_in[1];
    const int*   dst     = (const int*)  d_in[2];
    const int*   etype   = (const int*)  d_in[3];
    const float* norm    = (const float*)d_in[4];
    const float* weight  = (const float*)d_in[5];
    const float* w_comp  = (const float*)d_in[6];
    const float* bias    = (const float*)d_in[7];
    float* out = (float*)d_out;

    // Workspace layout (256B aligned regions)
    char* ws = (char*)d_ws;
    size_t off = 0;
    __bf16* h_bf = (__bf16*)(ws + off); off += (size_t)N_NODES * IN_FEAT * 2;   // 12.8 MB
    off = (off + 255) & ~(size_t)255;
    __bf16* wt   = (__bf16*)(ws + off); off += (size_t)CAT_COLS * IN_FEAT * 2;  // 256 KB
    off = (off + 255) & ~(size_t)255;
    __bf16* hb   = (__bf16*)(ws + off); off += (size_t)N_NODES * CAT_COLS * 2;  // 102.4 MB
    off = (off + 255) & ~(size_t)255;
    float*  acc  = (float*) (ws + off); off += (size_t)N_NODES * OUT_FEAT * 4;  // 25.6 MB

    // Phase 0: conversions + accumulator reset
    rgcn_convert_h<<<(N_NODES * IN_FEAT + 255) / 256, 256, 0, stream>>>(h, h_bf);
    rgcn_pack_w  <<<(CAT_COLS * IN_FEAT + 255) / 256, 256, 0, stream>>>(weight, wt);
    rgcn_zero_acc<<<(N_NODES * OUT_FEAT + 255) / 256, 256, 0, stream>>>(acc);

    // Phase 1: WMMA basis projection. Block = 8 m-tiles x 1 n-strip;
    // grid = 391 m-groups * 16 n-strips (tail m-tiles guarded in-kernel).
    rgcn_wmma_gemm<<<M_GROUPS * 16, 256, 0, stream>>>(h_bf, wt, hb);

    // Phase 2: per-edge combine + scatter (2 edges / block)
    rgcn_edge_scatter<<<N_EDGES / 2, 256, 0, stream>>>(hb, src, dst, etype, norm,
                                                       w_comp, acc);

    // Phase 3: bias + relu
    rgcn_finalize<<<(N_NODES * OUT_FEAT + 255) / 256, 256, 0, stream>>>(acc, bias, out);
}
